// Transformer_3152505995898
// MI455X (gfx1250) — matrix operand order repaired and checked by reading, not compiled, not run on hardware
//
#include <hip/hip_runtime.h>
#include <cstdint>
#include <cstddef>

// ---------------------------------------------------------------------------
// Types for CDNA5 WMMA (wave32): D(16x16 f32) = A(16x32 bf16) x B(32x16 bf16) + C
// ---------------------------------------------------------------------------
typedef __attribute__((ext_vector_type(16))) __bf16 v16bf;
typedef __attribute__((ext_vector_type(8)))  float  v8f;
typedef __attribute__((ext_vector_type(4)))  int    v4i;
typedef __attribute__((ext_vector_type(2)))  int    v2i;

union BF16Frag {
    v16bf v;
    unsigned short u[16];
    uint4 q[2];
};

static __device__ __forceinline__ v8f zero8() {
    v8f z = {0.f, 0.f, 0.f, 0.f, 0.f, 0.f, 0.f, 0.f};
    return z;
}

static __device__ __forceinline__ unsigned short f2bf(float f) {
    union { float f; unsigned u; } v; v.f = f;
    unsigned u = v.u;
    unsigned r = (u + 0x7FFFu + ((u >> 16) & 1u)) >> 16;   // round-to-nearest-even
    return (unsigned short)r;
}

static __device__ __forceinline__ v8f wmma_bf16(v16bf a, v16bf b, v8f c) {
    // (neg_a, A, neg_b, B, c_mod, C, reuse_a, reuse_b)
    return __builtin_amdgcn_wmma_f32_16x16x32_bf16(false, a, false, b, (short)0, c, false, false);
}

// ---------------------------------------------------------------------------
// Async global->LDS copies (CDNA5 GLOBAL_LOAD_ASYNC_TO_LDS_B*, ASYNCcnt),
// with a guaranteed-compiling fallback if the builtins are absent.
// Builtin signature (from clang diagnostics): (v4i AS1*, v4i AS3*, imm, imm)
// ---------------------------------------------------------------------------
#if __has_builtin(__builtin_amdgcn_global_load_async_to_lds_b128) && \
    __has_builtin(__builtin_amdgcn_global_load_async_to_lds_b64)  && \
    __has_builtin(__builtin_amdgcn_s_wait_asynccnt)
#define HAS_ASYNC_LDS 1
#else
#define HAS_ASYNC_LDS 0
#endif

static __device__ __forceinline__ void copy16_async(const unsigned short* g,
                                                    unsigned short* l) {
#if HAS_ASYNC_LDS
    __builtin_amdgcn_global_load_async_to_lds_b128(
        (__attribute__((address_space(1))) v4i*)g,
        (__attribute__((address_space(3))) v4i*)l, 0, 0);
#else
    *(uint4*)l = *(const uint4*)g;
#endif
}

static __device__ __forceinline__ void copy8_async(const unsigned short* g,
                                                   unsigned short* l) {
#if HAS_ASYNC_LDS
    __builtin_amdgcn_global_load_async_to_lds_b64(
        (__attribute__((address_space(1))) v2i*)g,
        (__attribute__((address_space(3))) v2i*)l, 0, 0);
#else
    *(uint2*)l = *(const uint2*)g;
#endif
}

static __device__ __forceinline__ void async_wait() {
#if HAS_ASYNC_LDS
    __builtin_amdgcn_s_wait_asynccnt(0);
#endif
}

// A fragment (16x32 bf16) from LDS tile stored [row][k], row stride in ushorts.
// Lanes 0-15: M=lane, elems 0..15 = K 0..15
// Lanes 16-31: M=lane-16, elems 0..15 = K 16..31
static __device__ __forceinline__ v16bf load_a_frag(const unsigned short* base,
                                                    int rowStride, int lane) {
    int m  = lane & 15;
    int ko = (lane < 16) ? 0 : 16;
    BF16Frag f;
    f.q[0] = *(const uint4*)(base + m * rowStride + ko);
    f.q[1] = *(const uint4*)(base + m * rowStride + 8 + ko);
    return f.v;
}

// B fragment (32x16 bf16): B[k][n]. Stored in LDS transposed: [n][k], col stride.
// Lanes 0-15: N=lane, elems 0..15 = K 0..15; Lanes 16-31: N=lane-16, K 16..31
static __device__ __forceinline__ v16bf load_b_frag(const unsigned short* base,
                                                    int colStride, int lane) {
    int nn = lane & 15;
    int kb = (lane < 16) ? 0 : 16;
    const uint4* p = (const uint4*)(base + nn * colStride + kb);
    BF16Frag f;
    f.q[0] = p[0];
    f.q[1] = p[1];
    return f.v;
}

// ---------------------------------------------------------------------------
// f32 -> bf16 bulk convert (n must be a multiple of 4)
// ---------------------------------------------------------------------------
__global__ void __launch_bounds__(256)
f2bf_kernel(const float* __restrict__ in, unsigned short* __restrict__ out, long long n4) {
    for (long long i = (long long)blockIdx.x * 256 + threadIdx.x; i < n4;
         i += (long long)gridDim.x * 256) {
        float4 v = ((const float4*)in)[i];
        unsigned lo = (unsigned)f2bf(v.x) | ((unsigned)f2bf(v.y) << 16);
        unsigned hi = (unsigned)f2bf(v.z) | ((unsigned)f2bf(v.w) << 16);
        ((uint2*)out)[i] = make_uint2(lo, hi);
    }
}

// ---------------------------------------------------------------------------
// Generic bf16 WMMA GEMM: C[M,Nc] = A[M,K] * W[K,Nc] + bias, tile 128x128x32.
// 8 waves, wave w owns a 16-row strip, 8 column subtiles (64 accum VGPRs).
// Double-buffered LDS: A tile via async global->LDS, W tile via reg staging.
// ---------------------------------------------------------------------------
enum { EP_BF16 = 0, EP_LEAKY_BF16 = 1, EP_RES_F32 = 2, EP_F32 = 3 };

template <int EP>
__global__ void __launch_bounds__(256)
gemm_kernel(const unsigned short* __restrict__ A,
            const unsigned short* __restrict__ W,
            const float* __restrict__ bias,
            const float* __restrict__ res,
            float* __restrict__ outF,
            unsigned short* __restrict__ outB,
            int M, int K, int Nc) {
    __shared__ unsigned short a_s[2][128 * 32];   // [row][k]
    __shared__ unsigned short w_s[2][128 * 32];   // [col][k] (transposed)

    const int tid  = threadIdx.x;
    const int lane = tid & 31;
    const int wid  = tid >> 5;
    const int m0   = blockIdx.y * 128;
    const int n0   = blockIdx.x * 128;

    v8f acc[8];
#pragma unroll
    for (int i = 0; i < 8; i++) acc[i] = zero8();

    const int arow = tid >> 1;           // 0..127
    const int asub = (tid & 1) * 16;     // 0 / 16
    const int wkk  = tid >> 3;           // 0..31
    const int wcg  = (tid & 7) * 16;     // col group of 16

    const unsigned short* aptr = A + (size_t)(m0 + arow) * K + asub;

    // --- prologue: stage tile for kt = 0 into buffer 0 ---
    uint4 tv[2];
    copy16_async(aptr, &a_s[0][arow * 32 + asub]);
    copy16_async(aptr + 8, &a_s[0][arow * 32 + asub + 8]);
    {
        const uint4* g = (const uint4*)(W + (size_t)wkk * Nc + n0 + wcg);
        tv[0] = g[0];
        tv[1] = g[1];
        const unsigned short* pv = (const unsigned short*)tv;
#pragma unroll
        for (int j = 0; j < 16; j++) w_s[0][(wcg + j) * 32 + wkk] = pv[j];
    }
    async_wait();
    __syncthreads();

    for (int kt = 0; kt < K; kt += 32) {
        const int buf   = (kt >> 5) & 1;
        const bool more = (kt + 32) < K;
        if (more) {
            // stream next tile while this one feeds the WMMAs
            copy16_async(aptr + kt + 32, &a_s[buf ^ 1][arow * 32 + asub]);
            copy16_async(aptr + kt + 40, &a_s[buf ^ 1][arow * 32 + asub + 8]);
            const uint4* g = (const uint4*)(W + (size_t)(kt + 32 + wkk) * Nc + n0 + wcg);
            tv[0] = g[0];
            tv[1] = g[1];
            const unsigned short* pv = (const unsigned short*)tv;
#pragma unroll
            for (int j = 0; j < 16; j++) w_s[buf ^ 1][(wcg + j) * 32 + wkk] = pv[j];
            if (kt + 64 < K)
                __builtin_prefetch(aptr + kt + 64, 0, 1);
        }
        v16bf af = load_a_frag(&a_s[buf][wid * 16 * 32], 32, lane);
#pragma unroll
        for (int n = 0; n < 8; n++) {
            v16bf bf = load_b_frag(&w_s[buf][n * 16 * 32], 32, lane);
            acc[n] = wmma_bf16(af, bf, acc[n]);
        }
        if (more) async_wait();
        __syncthreads();
    }

    // Epilogue. C/D layout: lanes 0-15: N=lane, M=r; lanes 16-31: N=lane-16, M=8+r
    const int mlo  = (lane < 16) ? 0 : 8;
    const int ncol = lane & 15;
#pragma unroll
    for (int n = 0; n < 8; n++) {
        int col = n0 + n * 16 + ncol;
        float b = bias[col];
#pragma unroll
        for (int r = 0; r < 8; r++) {
            int grow   = m0 + wid * 16 + mlo + r;
            size_t idx = (size_t)grow * Nc + col;
            float v    = acc[n][r] + b;
            if (EP == EP_LEAKY_BF16) {
                v = (v >= 0.f) ? v : 0.01f * v;
                outB[idx] = f2bf(v);
            } else if (EP == EP_BF16) {
                outB[idx] = f2bf(v);
            } else if (EP == EP_RES_F32) {
                outF[idx] = v + res[idx];
            } else {
                outF[idx] = v;
            }
        }
    }
}

// ---------------------------------------------------------------------------
// Attention: one block per (q-tile of 32 rows, head h, batch n).
// score pass computes a 32x128 contraction over E=256 (q.k^T or q.Erel^T).
// ---------------------------------------------------------------------------
static __device__ __forceinline__ void score_pass(
        const unsigned short* __restrict__ qg, size_t qbase,
        const unsigned short* __restrict__ sg, size_t sbase, int sstep,
        unsigned short* q_s, unsigned short* k_s, float* out, float scale,
        int tid, int lane, int wid) {
    v8f acc0 = zero8(), acc1 = zero8();
    const int qrow = tid >> 3, qoff = (tid & 7) * 4;   // q: 32 rows x 32 k
    const int krow = tid >> 1, koff = (tid & 1) * 16;  // k: 128 rows x 32 k
    const int rt = wid & 1, nc = (wid >> 1) * 2;

    for (int kt = 0; kt < 256; kt += 32) {
        copy8_async(qg + qbase + (size_t)qrow * 4096 + kt + qoff,
                    q_s + qrow * 32 + qoff);
        const unsigned short* g = sg + sbase + (size_t)krow * sstep + kt + koff;
        copy16_async(g, k_s + krow * 32 + koff);
        copy16_async(g + 8, k_s + krow * 32 + koff + 8);
        async_wait();
        __syncthreads();
        v16bf a  = load_a_frag(q_s + rt * 16 * 32, 32, lane);
        v16bf b0 = load_b_frag(k_s + nc * 16 * 32, 32, lane);
        v16bf b1 = load_b_frag(k_s + (nc + 1) * 16 * 32, 32, lane);
        acc0 = wmma_bf16(a, b0, acc0);
        acc1 = wmma_bf16(a, b1, acc1);
        __syncthreads();
    }
    const int mlo = (lane < 16) ? 0 : 8, ncol = lane & 15;
#pragma unroll
    for (int r = 0; r < 8; r++) {
        out[(rt * 16 + mlo + r) * 128 + nc * 16 + ncol]       = acc0[r] * scale;
        out[(rt * 16 + mlo + r) * 128 + (nc + 1) * 16 + ncol] = acc1[r] * scale;
    }
}

__global__ void __launch_bounds__(256)
attn_kernel(const unsigned short* __restrict__ qb,
            const unsigned short* __restrict__ kb,
            const unsigned short* __restrict__ vb,
            const unsigned short* __restrict__ eb,
            unsigned short* __restrict__ zb) {
    __shared__ unsigned short k_s[128 * 32];          // 8 KB: k / Erel stream
    __shared__ unsigned short q_s[32 * 32];           // 2 KB
    __shared__ float qe_s[32 * 128];                  // 16 KB
    __shared__ __align__(16) char sm[32 * 128 * 4];   // 16 KB: score f32, later v chunk
    __shared__ unsigned short att_s[32 * 128];        // 8 KB
    __shared__ float red[256];                        // 1 KB

    float* score = (float*)sm;
    unsigned short* v_s = (unsigned short*)sm;        // [col 64][k 128]

    const int tid = threadIdx.x, lane = tid & 31, wid = tid >> 5;
    const int q0 = blockIdx.x * 32;
    const int h  = blockIdx.y;
    const int n  = blockIdx.z;

    const size_t qbase = ((size_t)n * 2048 + (size_t)q0 * 16 + h) * 256;
    const size_t kbase = ((size_t)n * 2048 + h) * 256;
    const size_t ebase = (size_t)h * 128 * 256;

    // qk / sqrt(E) and QE
    score_pass(qb, qbase, kb, kbase, 4096, q_s, k_s, score, 0.0625f, tid, lane, wid);
    __syncthreads();
    score_pass(qb, qbase, eb, ebase, 256, q_s, k_s, qe_s, 1.0f, tid, lane, wid);
    __syncthreads();

    // Softmax + skew:  S[i][j] = (j <= i) ? QE[i][j + 127 - i] : 0
    {
        const int srow = tid >> 3, ssub = tid & 7;
        float* sr = score + srow * 128 + ssub * 16;
        float m = -1e30f;
#pragma unroll
        for (int j = 0; j < 16; j++) m = fmaxf(m, sr[j]);
        red[srow * 8 + ssub] = m;
        __syncthreads();
        float rm = red[srow * 8];
#pragma unroll
        for (int i = 1; i < 8; i++) rm = fmaxf(rm, red[srow * 8 + i]);
        float s = 0.f;
#pragma unroll
        for (int j = 0; j < 16; j++) {
            float e = __expf(sr[j] - rm);
            sr[j] = e;
            s += e;
        }
        __syncthreads();
        red[srow * 8 + ssub] = s;
        __syncthreads();
        float rs = 0.f;
#pragma unroll
        for (int i = 0; i < 8; i++) rs += red[srow * 8 + i];
        float inv = 1.f / rs;
        int ig = q0 + srow;
#pragma unroll
        for (int j = 0; j < 16; j++) {
            int jj = ssub * 16 + j;
            float att = sr[j] * inv;
            if (jj <= ig) att += qe_s[srow * 128 + (jj + 127 - ig)];
            att_s[srow * 128 + jj] = f2bf(att);
        }
    }

    // z = att(32x128) @ v(128x256), processed in 64-col chunks
    const int rt = wid & 1, nc = wid >> 1;
    const int mlo = (lane < 16) ? 0 : 8, ncol = lane & 15;
    const int vl = tid >> 1, vh = (tid & 1) * 32;
    for (int c0 = 0; c0 < 256; c0 += 64) {
        __syncthreads();   // score region is dead -> reuse as v_s
        {
            uint4 tv[4];
            const uint4* g = (const uint4*)(vb + ((size_t)n * 2048 + (size_t)vl * 16 + h) * 256 + c0 + vh);
            tv[0] = g[0]; tv[1] = g[1]; tv[2] = g[2]; tv[3] = g[3];
            const unsigned short* pv = (const unsigned short*)tv;
#pragma unroll
            for (int j = 0; j < 32; j++) v_s[(vh + j) * 128 + vl] = pv[j];
        }
        __syncthreads();
        v8f az = zero8();
#pragma unroll
        for (int ks = 0; ks < 4; ks++) {
            v16bf a = load_a_frag(att_s + rt * 16 * 128 + ks * 32, 128, lane);
            v16bf b = load_b_frag(v_s + nc * 16 * 128 + ks * 32, 128, lane);
            az = wmma_bf16(a, b, az);
        }
#pragma unroll
        for (int r = 0; r < 8; r++) {
            int l = q0 + rt * 16 + mlo + r;
            zb[((size_t)n * 2048 + (size_t)l * 16 + h) * 256 + c0 + nc * 16 + ncol] = f2bf(az[r]);
        }
    }
}

// ---------------------------------------------------------------------------
// Batchnorm over axes (0,2): channel c = row % 2048, stats over 32 rows x 256
// ---------------------------------------------------------------------------
__global__ void __launch_bounds__(256)
bn_stats_kernel(const float* __restrict__ t, const float* __restrict__ g,
                const float* __restrict__ be, float* __restrict__ A,
                float* __restrict__ B) {
    const int c = blockIdx.x, tid = threadIdx.x;
    float s = 0.f, sq = 0.f;
    for (int nb = 0; nb < 32; nb++) {
        float v = t[((size_t)nb * 2048 + c) * 256 + tid];
        s += v;
        sq += v * v;
    }
    __shared__ float rs[256], rq[256];
    rs[tid] = s; rq[tid] = sq;
    __syncthreads();
    for (int st = 128; st > 0; st >>= 1) {
        if (tid < st) { rs[tid] += rs[tid + st]; rq[tid] += rq[tid + st]; }
        __syncthreads();
    }
    if (tid == 0) {
        float mean = rs[0] * (1.f / 8192.f);
        float var  = rq[0] * (1.f / 8192.f) - mean * mean;
        float r    = rsqrtf(var + 1e-5f);
        float a    = r * g[c];
        A[c] = a;
        B[c] = be[c] - mean * a;
    }
}

template <bool WF, bool WB>
__global__ void __launch_bounds__(256)
bn_apply_kernel(const float* __restrict__ t, const float* __restrict__ A,
                const float* __restrict__ B, float* __restrict__ outF,
                unsigned short* __restrict__ outB) {
    const size_t total = (size_t)65536 * 64;   // float4 count
    for (size_t i = (size_t)blockIdx.x * 256 + threadIdx.x; i < total;
         i += (size_t)gridDim.x * 256) {
        size_t row = i >> 6;
        int c = (int)(row & 2047);
        float4 v = ((const float4*)t)[i];
        float a = A[c], b = B[c];
        v.x = v.x * a + b; v.y = v.y * a + b; v.z = v.z * a + b; v.w = v.w * a + b;
        if (WF) ((float4*)outF)[i] = v;
        if (WB) {
            unsigned lo = (unsigned)f2bf(v.x) | ((unsigned)f2bf(v.y) << 16);
            unsigned hi = (unsigned)f2bf(v.z) | ((unsigned)f2bf(v.w) << 16);
            ((uint2*)outB)[i] = make_uint2(lo, hi);
        }
    }
}

// ---------------------------------------------------------------------------
// Host-side orchestration
// ---------------------------------------------------------------------------
extern "C" void kernel_launch(void* const* d_in, const int* in_sizes, int n_in,
                              void* d_out, int out_size, void* d_ws, size_t ws_size,
                              hipStream_t stream) {
    (void)in_sizes; (void)n_in; (void)out_size; (void)ws_size;

    const float* x    = (const float*)d_in[0];
    const float* Wq   = (const float*)d_in[1];
    const float* bq   = (const float*)d_in[2];
    const float* Wk   = (const float*)d_in[3];
    const float* bk   = (const float*)d_in[4];
    const float* Wv   = (const float*)d_in[5];
    const float* bv   = (const float*)d_in[6];
    const float* Erel = (const float*)d_in[7];
    const float* Wo   = (const float*)d_in[8];
    const float* bo   = (const float*)d_in[9];
    const float* g1   = (const float*)d_in[10];
    const float* be1  = (const float*)d_in[11];
    const float* g2   = (const float*)d_in[12];
    const float* be2  = (const float*)d_in[13];
    const float* W1   = (const float*)d_in[14];
    const float* b1   = (const float*)d_in[15];
    const float* W2   = (const float*)d_in[16];
    const float* b2   = (const float*)d_in[17];
    const float* Wenc = (const float*)d_in[18];
    const float* benc = (const float*)d_in[19];

    char* w = (char*)d_ws;
    const size_t MB = 1ull << 20;
    // region 0 (128 MB): q/k/v bf16 (32 MB each), later reused for FFN hidden a1
    unsigned short* q_bf  = (unsigned short*)(w + 0 * MB);
    unsigned short* k_bf  = (unsigned short*)(w + 32 * MB);
    unsigned short* v_bf  = (unsigned short*)(w + 64 * MB);
    unsigned short* a1_bf = (unsigned short*)(w + 0 * MB);
    // region 1 (32 MB): xs_bf -> z_bf -> h1_bf -> out2_bf
    unsigned short* actB = (unsigned short*)(w + 128 * MB);
    // region 2/3: f32 temporaries
    float* tbuf = (float*)(w + 160 * MB);
    float* h1f  = (float*)(w + 224 * MB);
    // weights (bf16)
    char* wts = w + 288 * MB;
    unsigned short* wq_bf   = (unsigned short*)(wts + 0);
    unsigned short* wk_bf   = (unsigned short*)(wts + 131072);
    unsigned short* wv_bf   = (unsigned short*)(wts + 262144);
    unsigned short* wo_bf   = (unsigned short*)(wts + 393216);
    unsigned short* wenc_bf = (unsigned short*)(wts + 524288);
    unsigned short* w1_bf   = (unsigned short*)(wts + 655360);
    unsigned short* w2_bf   = (unsigned short*)(wts + 1179648);
    unsigned short* erel_bf = (unsigned short*)(wts + 1703936);
    // BN per-channel scale/shift
    float* bnA1 = (float*)(w + 292 * MB);
    float* bnB1 = (float*)(w + 292 * MB + 8192);
    float* bnA2 = (float*)(w + 292 * MB + 16384);
    float* bnB2 = (float*)(w + 292 * MB + 24576);

    auto cvt = [&](const float* src, unsigned short* dst, size_t nelem) {
        long long n4 = (long long)(nelem / 4);
        int blocks = (int)((n4 + 255) / 256);
        if (blocks > 4096) blocks = 4096;
        f2bf_kernel<<<blocks, 256, 0, stream>>>(src, dst, n4);
    };

    cvt(x, actB, (size_t)65536 * 256);
    cvt(Wq, wq_bf, 65536);
    cvt(Wk, wk_bf, 65536);
    cvt(Wv, wv_bf, 65536);
    cvt(Wo, wo_bf, 65536);
    cvt(Wenc, wenc_bf, 65536);
    cvt(W1, w1_bf, 262144);
    cvt(W2, w2_bf, 262144);
    cvt(Erel, erel_bf, 524288);

    dim3 blk(256);
    dim3 g256(2, 512);

    // q, k, v projections (store bf16)
    gemm_kernel<EP_BF16><<<g256, blk, 0, stream>>>(actB, wq_bf, bq, nullptr, nullptr, q_bf, 65536, 256, 256);
    gemm_kernel<EP_BF16><<<g256, blk, 0, stream>>>(actB, wk_bf, bk, nullptr, nullptr, k_bf, 65536, 256, 256);
    gemm_kernel<EP_BF16><<<g256, blk, 0, stream>>>(actB, wv_bf, bv, nullptr, nullptr, v_bf, 65536, 256, 256);

    // attention: z (bf16) overwrites actB (xs_bf now dead)
    attn_kernel<<<dim3(4, 16, 32), blk, 0, stream>>>(q_bf, k_bf, v_bf, erel_bf, actB);

    // t1 = z@Wo + bo + xs
    gemm_kernel<EP_RES_F32><<<g256, blk, 0, stream>>>(actB, wo_bf, bo, x, tbuf, nullptr, 65536, 256, 256);

    // h1 = BN1(t1): f32 + bf16 copies
    bn_stats_kernel<<<2048, 256, 0, stream>>>(tbuf, g1, be1, bnA1, bnB1);
    bn_apply_kernel<true, true><<<4096, 256, 0, stream>>>(tbuf, bnA1, bnB1, h1f, actB);

    // a1 = leaky(h1@W1 + b1)  (bf16, 65536x1024)
    gemm_kernel<EP_LEAKY_BF16><<<dim3(8, 512), blk, 0, stream>>>(actB, w1_bf, b1, nullptr, nullptr, a1_bf, 65536, 256, 1024);

    // t2 = a1@W2 + b2 + h1
    gemm_kernel<EP_RES_F32><<<g256, blk, 0, stream>>>(a1_bf, w2_bf, b2, h1f, tbuf, nullptr, 65536, 1024, 256);

    // out2 = BN2(t2) (bf16 only)
    bn_stats_kernel<<<2048, 256, 0, stream>>>(tbuf, g2, be2, bnA2, bnB2);
    bn_apply_kernel<false, true><<<4096, 256, 0, stream>>>(tbuf, bnA2, bnB2, nullptr, actB);

    // final: out2@Wenc + benc -> d_out (f32)
    gemm_kernel<EP_F32><<<g256, blk, 0, stream>>>(actB, wenc_bf, benc, nullptr, (float*)d_out, nullptr, 65536, 256, 256);
}
